// AudioVQVAE_52544629899305
// MI455X (gfx1250) — matrix operand order repaired and checked by reading, not compiled, not run on hardware
//
#include <hip/hip_runtime.h>
#include <hip/hip_bf16.h>

// AudioVQVAE forward for gfx1250 (MI455X): all 128-channel contractions mapped
// to v_wmma_f32_16x16x32_f16 tiles (wave32). Weights are pre-converted/packed
// to f16 so A-fragments load as contiguous b128 runs; fragment loads are
// double-buffered against the WMMA pipe. Intermediates live in d_ws and are
// L2-resident (<< 192MB); fp32 is kept only where the losses / argmax need it.

typedef __attribute__((ext_vector_type(16))) _Float16 v16h;
typedef __attribute__((ext_vector_type(8)))  float    v8f;

constexpr int CB  = 4;        // batch
constexpr int CT  = 131072;   // waveform length
constexpr int CT1 = 32768;    // after conv1 (stride 4)
constexpr int CTQ = 8192;     // after conv2 (stride 4)
constexpr int CD  = 128;      // channels / code dim
constexpr int CK  = 2048;     // codebook size

__device__ __forceinline__ float gelu_f(float v) {
    return 0.5f * v * (1.0f + erff(v * 0.70710678118654752f));
}

// blockDim.x must be 256 (8 waves). Returns block-wide sum to all threads.
__device__ __forceinline__ float block_sum256(float v) {
    __shared__ float red[8];
    const int lane = threadIdx.x & 31, wid = threadIdx.x >> 5;
#pragma unroll
    for (int off = 16; off >= 1; off >>= 1) v += __shfl_xor(v, off, 32);
    __syncthreads();                       // protect red[] across repeated calls
    if (lane == 0) red[wid] = v;
    __syncthreads();
    float s = red[0];
#pragma unroll
    for (int i = 1; i < 8; ++i) s += red[i];
    return s;
}

__global__ void zero_acc(float* a) {
    if (threadIdx.x < 2) a[threadIdx.x] = 0.f;
}

// ---- elementwise f32 -> f16 (weight pre-conversion) ----
__global__ void cvt_f32_f16(const float* __restrict__ s, _Float16* __restrict__ d, int n) {
    const int i = blockIdx.x * 256 + threadIdx.x;
    if (i < n) d[i] = (_Float16)s[i];
}

// ---- pack ConvTranspose1d(128->128,k=8,s=4,p=2) weights per output phase:
// wpt[r][co][ci*2+s] = wt1[ci][co][tap(r,s)], tap(r,0)=(r+2)&3, tap(r,1)=tap+4
__global__ void pack_wt1(const float* __restrict__ wt1, _Float16* __restrict__ wpt) {
    const int id = blockIdx.x * 256 + threadIdx.x;   // 4*128*256
    const int kk = id & 255;
    const int co = (id >> 8) & 127;
    const int r  = id >> 15;
    const int ci = kk >> 1, sl = kk & 1;
    const int k0 = (r + 2) & 3;
    wpt[id] = (_Float16)wt1[(ci * CD + co) * 8 + (sl ? (k0 + 4) : k0)];
}

// ---- conv1: 1 -> 128 ch, k=8, stride 4, pad 2, GELU. Scalar (Cin=1). ----
__global__ void conv1_gelu(const float* __restrict__ x, const float* __restrict__ w1,
                           const float* __restrict__ b1, _Float16* __restrict__ z1) {
    const int id = blockIdx.x * 256 + threadIdx.x;     // CB*CD*CT1
    const int t  = id % CT1;
    const int co = (id / CT1) & (CD - 1);
    const int b  = id / (CT1 * CD);
    float acc = b1[co];
    const float* xb = x + (long)b * CT;
#pragma unroll
    for (int j = 0; j < 8; ++j) {
        const int pos = 4 * t - 2 + j;
        if (pos >= 0 && pos < CT) acc += xb[pos] * w1[co * 8 + j];
    }
    z1[id] = (_Float16)gelu_f(acc);
}

// ---- generic conv-as-GEMM via WMMA: out[co,t] = sum_{ci,j} w[co,ci,j]*in[ci,S*t-P+j]
// One 16(co) x 16(t) tile per wave; K_eff = 128*KT in chunks of 32, double-buffered.
template<int KT, int STRIDE, int PAD, bool DOGELU>
__global__ void conv_gemm_wmma(const _Float16* __restrict__ in,
                               const _Float16* __restrict__ wh,   // [128][128*KT] f16
                               const float* __restrict__ bias,
                               _Float16* __restrict__ out16,      // nullable
                               float* __restrict__ out32,         // nullable
                               int Lin, int Lout) {
    const int lane  = threadIdx.x & 31;
    const int wave  = threadIdx.x >> 5;
    const int ntile = Lout >> 4;
    const int tile  = blockIdx.x * 8 + wave;
    const int nt    = tile % ntile;
    const int ct    = (tile / ntile) & 7;
    const int b     = tile / (ntile * 8);
    const int m     = lane & 15;
    const int half  = lane >> 4;
    const int akb   = half * 8;
    const int co_row = ct * 16 + m;   // A row (M)
    const int tcol   = nt * 16 + m;   // B col (N), same lane formula
    constexpr int KEFF = CD * KT;

    const _Float16* wrow = wh + (long)co_row * KEFF;
    const _Float16* inb  = in + (long)b * CD * Lin;

    auto loadAB = [&](int kc, v16h& A, v16h& Bv) {
#pragma unroll
        for (int j = 0; j < 16; ++j)          // A: two contiguous 8-half runs
            A[j] = wrow[kc + akb + ((j < 8) ? j : (j + 8))];
#pragma unroll
        for (int j = 0; j < 16; ++j) {        // B: 32x16 f16 layout (im2col)
            const int kk = kc + akb + ((j < 8) ? j : (j + 8));
            const int ci = kk / KT, jt = kk % KT;
            const int pos = STRIDE * tcol - PAD + jt;
            _Float16 v = (_Float16)0.f;
            if (pos >= 0 && pos < Lin) v = inb[(long)ci * Lin + pos];
            Bv[j] = v;
        }
    };

    v8f acc = {};
    v16h ta, tb, na, nb;
    loadAB(0, ta, tb);
    for (int kc = 0; kc < KEFF; kc += 32) {
        if (kc + 32 < KEFF) {
            // prefetch the next chunk's activation row toward the WGP
            const int cin = (kc + 32 + half * 16) / KT;
            int pp = STRIDE * tcol - PAD; if (pp < 0) pp = 0;
            __builtin_prefetch((const void*)&inb[(long)cin * Lin + pp], 0, 3);
            loadAB(kc + 32, na, nb);
        }
        acc = __builtin_amdgcn_wmma_f32_16x16x32_f16(false, ta, false, tb,
                                                     (short)0, acc, false, false);
        ta = na; tb = nb;
    }
#pragma unroll
    for (int i = 0; i < 8; ++i) {             // C/D: VGPR i <-> M = i + half*8
        const int co = ct * 16 + i + half * 8;
        const int t  = nt * 16 + (lane & 15);
        float v = acc[i] + bias[co];
        if (DOGELU) v = gelu_f(v);
        const long o = ((long)b * CD + co) * Lout + t;
        if (out16) out16[o] = (_Float16)v;
        if (out32) out32[o] = v;
    }
}

// ---- normalize codebook rows -> f16 ----
__global__ void emb_normalize(const float* __restrict__ emb, _Float16* __restrict__ embn) {
    const int row = blockIdx.x;       // CK
    const int tid = threadIdx.x;      // 128
    const float v = emb[row * CD + tid];
    float s = v * v;
#pragma unroll
    for (int off = 16; off >= 1; off >>= 1) s += __shfl_xor(s, off, 32);
    __shared__ float red[4];
    if ((tid & 31) == 0) red[tid >> 5] = s;
    __syncthreads();
    const float tot = red[0] + red[1] + red[2] + red[3];
    const float invn = 1.f / fmaxf(sqrtf(tot), 1e-12f);
    embn[row * CD + tid] = (_Float16)(v * invn);
}

// ---- VQ: cosine-sim argmax via WMMA. One wave = 16 time rows vs all 2048 codes.
__global__ void vq_argmax_wmma(const float* __restrict__ zf,
                               const _Float16* __restrict__ embn,
                               int* __restrict__ idxb) {
    const int lane = threadIdx.x & 31;
    const int wave = threadIdx.x >> 5;
    const int wg   = blockIdx.x * 8 + wave;          // CB*(CTQ/16) waves
    const int b    = wg / (CTQ / 16);
    const int t0   = (wg % (CTQ / 16)) * 16;
    const int m    = lane & 15;
    const int half = lane >> 4;

    // per-row inverse L2 norm (2 lanes cooperate per row over 128 channels)
    const float* zrow = zf + (long)b * CD * CTQ + t0 + m;
    float s = 0.f;
    for (int ci = half * 64; ci < half * 64 + 64; ++ci) {
        const float v = zrow[(long)ci * CTQ];
        s += v * v;
    }
    s += __shfl_xor(s, 16, 32);
    const float invn = 1.f / fmaxf(sqrtf(s), 1e-12f);

    // cache the four A fragments (D=128 = 4 x K32)
    v16h afrag[4];
#pragma unroll
    for (int kc = 0; kc < 4; ++kc) {
#pragma unroll
        for (int j = 0; j < 16; ++j) {
            const int v  = j >> 1;
            const int ak = half * 16 + j + 0 * v;
            const int ci = kc * 32 + ak;
            afrag[kc][j] = (_Float16)(zrow[(long)ci * CTQ] * invn);
        }
    }

    float best[8]; int bidx[8];
#pragma unroll
    for (int i = 0; i < 8; ++i) { best[i] = -1e30f; bidx[i] = 0; }

    for (int ntv = 0; ntv < CK / 16; ++ntv) {
        const int code = ntv * 16 + (lane & 15);
        const _Float16* erow = embn + (long)code * CD + half * 16;
        // batch all four B-fragment loads, then run four back-to-back WMMAs
        v16h bf[4];
#pragma unroll
        for (int kc = 0; kc < 4; ++kc)
#pragma unroll
            for (int j = 0; j < 16; ++j)
                bf[kc][j] = erow[kc * 32 + j];
        v8f acc = {};
#pragma unroll
        for (int kc = 0; kc < 4; ++kc)
            acc = __builtin_amdgcn_wmma_f32_16x16x32_f16(false, afrag[kc], false, bf[kc],
                                                         (short)0, acc, false, false);
#pragma unroll
        for (int i = 0; i < 8; ++i)
            if (acc[i] > best[i]) { best[i] = acc[i]; bidx[i] = code; }
    }
    // butterfly argmax across the 16 lanes of each half (first-index ties)
#pragma unroll
    for (int off = 8; off >= 1; off >>= 1) {
#pragma unroll
        for (int i = 0; i < 8; ++i) {
            const float ov = __shfl_xor(best[i], off, 32);
            const int   oi = __shfl_xor(bidx[i], off, 32);
            if (ov > best[i] || (ov == best[i] && oi < bidx[i])) { best[i] = ov; bidx[i] = oi; }
        }
    }
    if ((lane & 15) == 0) {
#pragma unroll
        for (int i = 0; i < 8; ++i)
            idxb[b * CTQ + t0 + i + half * 8] = bidx[i];
    }
}

// ---- z_q gather (f16 for decoder), commitment-loss partial, idx -> d_out ----
__global__ void vq_gather_com(const float* __restrict__ zf,
                              const float* __restrict__ emb,
                              const int* __restrict__ idxb,
                              _Float16* __restrict__ zqh,
                              float* __restrict__ idx_out,
                              float* __restrict__ com_acc) {
    const int id = blockIdx.x * 256 + threadIdx.x;   // CB*CD*CTQ
    const int t  = id % CTQ;
    const int ci = (id / CTQ) & (CD - 1);
    const int b  = id / (CTQ * CD);
    const int code = idxb[b * CTQ + t];
    const float e  = emb[code * CD + ci];
    zqh[id] = (_Float16)e;
    if (ci == 0) idx_out[b * CTQ + t] = (float)code;
    const float d = zf[id] - e;
    const float bs = block_sum256(d * d);
    if (threadIdx.x == 0) atomicAdd(com_acc, bs);
}

// ---- convT1: 128->128, k=8, stride 4, pad 2, GELU. Phase-decomposed WMMA GEMM.
// out[co,4q+r] = sum_ci in[ci,q+o0]*w[ci,co,k0(r)] + in[ci,q+o0-1]*w[ci,co,k0(r)+4]
__global__ void convT1_wmma(const _Float16* __restrict__ h1,   // [B,128,CTQ]
                            const _Float16* __restrict__ wpt,  // [4][128][256] packed
                            const float* __restrict__ bt1,
                            _Float16* __restrict__ h2) {       // [B,128,CT1]
    const int lane = threadIdx.x & 31;
    const int wave = threadIdx.x >> 5;
    const int tile = blockIdx.x * 8 + wave;          // CB*4*8*(CTQ/16)
    const int qt = tile % (CTQ / 16);
    const int ct = (tile / (CTQ / 16)) & 7;
    const int r  = (tile / ((CTQ / 16) * 8)) & 3;
    const int b  = tile / ((CTQ / 16) * 8 * 4);
    const int m    = lane & 15;
    const int half = lane >> 4;
    const int akb  = half * 8;
    const int o0 = (r >= 2) ? 1 : 0, o1 = o0 - 1;
    const int co_row = ct * 16 + m;
    const int qcol   = qt * 16 + m;

    const _Float16* wrow = wpt + ((long)r * CD + co_row) * 256;
    const _Float16* inb  = h1 + (long)b * CD * CTQ;

    auto loadAB = [&](int kc, v16h& A, v16h& Bv) {
#pragma unroll
        for (int j = 0; j < 16; ++j)          // A: two contiguous 8-half runs
            A[j] = wrow[kc + akb + ((j < 8) ? j : (j + 8))];
#pragma unroll
        for (int j = 0; j < 16; ++j) {
            const int kk = kc + akb + ((j < 8) ? j : (j + 8));
            const int ci = kk >> 1, sl = kk & 1;
            const int pos = qcol + (sl ? o1 : o0);
            _Float16 v = (_Float16)0.f;
            if (pos >= 0 && pos < CTQ) v = inb[(long)ci * CTQ + pos];
            Bv[j] = v;
        }
    };

    v8f acc = {};
    v16h ta, tb, na, nb;
    loadAB(0, ta, tb);
    for (int kc = 0; kc < 256; kc += 32) {    // K_eff = 2 taps * 128 ch
        if (kc + 32 < 256) {
            const int cin = (kc + 32 + half * 16) >> 1;
            int pp = qcol + o1; if (pp < 0) pp = 0;
            __builtin_prefetch((const void*)&inb[(long)cin * CTQ + pp], 0, 3);
            loadAB(kc + 32, na, nb);
        }
        acc = __builtin_amdgcn_wmma_f32_16x16x32_f16(false, ta, false, tb,
                                                     (short)0, acc, false, false);
        ta = na; tb = nb;
    }
#pragma unroll
    for (int i = 0; i < 8; ++i) {
        const int co = ct * 16 + i + half * 8;
        const int q  = qt * 16 + (lane & 15);
        float v = gelu_f(acc[i] + bt1[co]);
        h2[((long)b * CD + co) * CT1 + (4 * q + r)] = (_Float16)v;
    }
}

// ---- convT2: 128->1, k=8, stride 4, pad 2, tanh; fused reconstruction loss ----
__global__ void convT2_tanh_rec(const _Float16* __restrict__ h2,  // [B,128,CT1]
                                const float* __restrict__ wt2,    // [128,1,8]
                                const float* __restrict__ bt2,
                                const float* __restrict__ x,
                                float* __restrict__ xr_out,
                                float* __restrict__ rec_acc) {
    const int id = blockIdx.x * 256 + threadIdx.x;   // CB*CT
    const int t  = id % CT;
    const int b  = id / CT;
    const int r  = t & 3, q = t >> 2;
    const int k0 = (r + 2) & 3, k1 = k0 + 4;
    const int p0 = q + ((r >= 2) ? 1 : 0);
    const int p1 = p0 - 1;
    const bool v0 = (p0 >= 0 && p0 < CT1);
    const bool v1 = (p1 >= 0 && p1 < CT1);
    float acc = bt2[0];
    const _Float16* hb = h2 + (long)b * CD * CT1;
    for (int ci = 0; ci < CD; ++ci) {
        const _Float16* hc = hb + (long)ci * CT1;
        if (v0) acc += (float)hc[p0] * wt2[ci * 8 + k0];
        if (v1) acc += (float)hc[p1] * wt2[ci * 8 + k1];
    }
    const float xr = tanhf(acc);
    xr_out[id] = xr;
    const float d = xr - x[id];
    const float bs = block_sum256(d * d);
    if (threadIdx.x == 0) atomicAdd(rec_acc, bs);
}

// ---- scalar losses: rec, com (from accumulators) + codebook perplexity ----
__global__ void finalize_losses(const float* __restrict__ cs,
                                const float* __restrict__ acc,
                                float* __restrict__ out) {
    float s = 0.f;
    for (int i = threadIdx.x; i < CK; i += 256) s += cs[i];
    const float tot = block_sum256(s) + 1e-6f;
    float h = 0.f;
    for (int i = threadIdx.x; i < CK; i += 256) {
        const float n = cs[i] / tot;
        h += n * logf(n + 1e-6f);
    }
    const float hs = block_sum256(h);
    if (threadIdx.x == 0) {
        out[0] = acc[1] / (float)((long)CB * CT);          // rec mse
        out[1] = acc[0] / (float)((long)CB * CD * CTQ);    // commitment mse
        out[2] = expf(-hs);                                // perplexity
    }
}

extern "C" void kernel_launch(void* const* d_in, const int* in_sizes, int n_in,
                              void* d_out, int out_size, void* d_ws, size_t ws_size,
                              hipStream_t stream) {
    (void)in_sizes; (void)n_in; (void)out_size; (void)ws_size;
    const float* x    = (const float*)d_in[0];
    const float* ew1  = (const float*)d_in[1];
    const float* eb1  = (const float*)d_in[2];
    const float* ew2  = (const float*)d_in[3];
    const float* eb2  = (const float*)d_in[4];
    const float* ew3  = (const float*)d_in[5];
    const float* eb3  = (const float*)d_in[6];
    const float* dw0  = (const float*)d_in[7];
    const float* db0  = (const float*)d_in[8];
    const float* dwt1 = (const float*)d_in[9];
    const float* dbt1 = (const float*)d_in[10];
    const float* dwt2 = (const float*)d_in[11];
    const float* dbt2 = (const float*)d_in[12];
    const float* emb  = (const float*)d_in[13];
    const float* csz  = (const float*)d_in[14];

    // Workspace layout (lifetime-based aliasing, ~60.1 MB total):
    char* ws = (char*)d_ws;
    size_t o = 0;
    _Float16* z1h = (_Float16*)(ws + o);   // conv1 out [B,128,CT1]  (f16)
    _Float16* h2h = z1h;                   // alias: convT1 out (z1h dead by then)
    o += (size_t)CB * CD * CT1 * 2;
    _Float16* z2h = (_Float16*)(ws + o);   // conv2 out [B,128,CTQ]  (f16)
    _Float16* zqh = z2h;                   // alias: z_q (z2h dead after conv3)
    o += (size_t)CB * CD * CTQ * 2;
    float* zf = (float*)(ws + o);          // conv3 out z (f32, for norm/com)
    _Float16* h1h = (_Float16*)zf;         // alias: dec0 out (zf dead after gather)
    o += (size_t)CB * CD * CTQ * 4;
    _Float16* embnh = (_Float16*)(ws + o); // normalized codebook f16
    o += (size_t)CK * CD * 2;
    int* idxb = (int*)(ws + o);            // argmax indices
    o += (size_t)CB * CTQ * 4;
    _Float16* w2h = (_Float16*)(ws + o);   // conv2 weights f16 [128][1024]
    o += (size_t)CD * CD * 8 * 2;
    _Float16* w3h = (_Float16*)(ws + o);   // conv3 weights f16 [128][384]
    o += (size_t)CD * CD * 3 * 2;
    _Float16* w0h = (_Float16*)(ws + o);   // dec0 weights f16 [128][384]
    o += (size_t)CD * CD * 3 * 2;
    _Float16* wpt = (_Float16*)(ws + o);   // convT1 packed f16 [4][128][256]
    o += (size_t)4 * CD * CD * 2 * 2;
    float* accv = (float*)(ws + o);        // accv[0]=com sum, accv[1]=rec sum

    float* out     = (float*)d_out;        // [rec, com, ppx, xr(B*T), idx(B*TQ)]
    float* xr_out  = out + 3;
    float* idx_out = out + 3 + (size_t)CB * CT;

    zero_acc<<<1, 32, 0, stream>>>(accv);
    // weight pre-conversion / packing (tiny, one-shot per call)
    cvt_f32_f16<<<(CD * CD * 8 + 255) / 256, 256, 0, stream>>>(ew2, w2h, CD * CD * 8);
    cvt_f32_f16<<<(CD * CD * 3 + 255) / 256, 256, 0, stream>>>(ew3, w3h, CD * CD * 3);
    cvt_f32_f16<<<(CD * CD * 3 + 255) / 256, 256, 0, stream>>>(dw0, w0h, CD * CD * 3);
    pack_wt1<<<(4 * CD * CD * 2) / 256, 256, 0, stream>>>(dwt1, wpt);
    emb_normalize<<<CK, 128, 0, stream>>>(emb, embnh);

    conv1_gelu<<<(CB * CD * CT1) / 256, 256, 0, stream>>>(x, ew1, eb1, z1h);
    conv_gemm_wmma<8, 4, 2, true ><<<(CB * 8 * (CTQ / 16)) / 8, 256, 0, stream>>>(
        z1h, w2h, eb2, z2h, nullptr, CT1, CTQ);
    conv_gemm_wmma<3, 1, 1, false><<<(CB * 8 * (CTQ / 16)) / 8, 256, 0, stream>>>(
        z2h, w3h, eb3, nullptr, zf, CTQ, CTQ);
    vq_argmax_wmma<<<(CB * (CTQ / 16)) / 8, 256, 0, stream>>>(zf, embnh, idxb);
    vq_gather_com<<<(CB * CD * CTQ) / 256, 256, 0, stream>>>(
        zf, emb, idxb, zqh, idx_out, accv);
    conv_gemm_wmma<3, 1, 1, true ><<<(CB * 8 * (CTQ / 16)) / 8, 256, 0, stream>>>(
        zqh, w0h, db0, h1h, nullptr, CTQ, CTQ);
    convT1_wmma<<<(CB * 4 * 8 * (CTQ / 16)) / 8, 256, 0, stream>>>(h1h, wpt, dbt1, h2h);
    convT2_tanh_rec<<<(CB * CT) / 256, 256, 0, stream>>>(
        h2h, dwt2, dbt2, x, xr_out, accv + 1);
    finalize_losses<<<1, 256, 0, stream>>>(csz, accv, out);
}
